// DemodConv2d_63660005262106
// MI455X (gfx1250) — matrix operand order repaired and checked by reading, not compiled, not run on hardware
//
#include <hip/hip_runtime.h>

// ---- CDNA5 vector types for WMMA ----
typedef __attribute__((ext_vector_type(16))) __bf16    v16bf;
typedef __attribute__((ext_vector_type(8)))  float     v8f;
typedef __attribute__((ext_vector_type(4)))  unsigned  v4u;

union FragU { v4u q[2]; v16bf v; };   // 32B: 8 VGPRs of packed bf16

// pack the high 16 bits (truncated bf16) of two fp32 words: {hi16(u1),hi16(u0)}
__device__ __forceinline__ unsigned pack_hi16(unsigned u1, unsigned u0) {
    return __builtin_amdgcn_perm(u1, u0, 0x07060302u);   // one v_perm_b32
}

// ---------------------------------------------------------------------------
// Kernel 1: demod[co] = rsqrt(sum_{ci,kh,kw} w^2 + 1e-8), 512 values -> d_ws
// ---------------------------------------------------------------------------
__global__ __launch_bounds__(256) void demod512_kernel(const float* __restrict__ w,
                                                       float* __restrict__ dm) {
    const int co = blockIdx.x;            // 0..511
    float s = 0.f;
    for (int i = threadIdx.x; i < 4608; i += 256) {
        float v = w[(size_t)co * 4608 + i];
        s += v * v;
    }
    __shared__ float red[256];
    red[threadIdx.x] = s;
    __syncthreads();
    for (int off = 128; off > 0; off >>= 1) {
        if (threadIdx.x < off) red[threadIdx.x] += red[threadIdx.x + off];
        __syncthreads();
    }
    if (threadIdx.x == 0) dm[co] = rsqrtf(red[0] + 1e-8f);
}

// ---------------------------------------------------------------------------
// Kernel 2: implicit-GEMM 3x3 conv, split-bf16 (hi/lo) WMMA 16x16x32,
// double-buffered LDS software pipeline, one barrier per K-step.
//   GEMM: M=Cout=512, N=H*W=65536, K=Cin*9=4608 (tap-major, 32-ci slices)
//   Block tile 128(M) x 128(N); 8 waves, each 32x64 = 2x4 16x16 frags.
// ---------------------------------------------------------------------------
__global__ __launch_bounds__(256) void demodconv_wmma_kernel(
    const float* __restrict__ in,    // (512,256,256)
    const float* __restrict__ wgt,   // (512,512,3,3)
    const float* __restrict__ demod, // (512)
    float* __restrict__ out)         // (512,256,256)
{
    constexpr int W  = 256;
    constexpr int HW = 256 * 256;
    constexpr int CIN = 512;
    constexpr int LDSTR = 40;        // shorts per LDS row (80B: 16B-aligned, bank-staggered)
    constexpr int NSTEP = 144;       // 9 taps * 16 ci-slices

    // [buf][hi/lo][row*LDSTR]  -> 2*2*128*40*2B = 40KB each, 80KB total
    __shared__ unsigned short ldsA[2][2][128 * LDSTR];  // A: [co][ci]
    __shared__ unsigned short ldsB[2][2][128 * LDSTR];  // B: [x ][ci]

    const int tid   = threadIdx.x;
    const int lane  = tid & 31;
    const int wv    = tid >> 5;
    const int mwave = wv >> 1;                // 0..3
    const int nwave = wv & 1;                 // 0..1
    const int x0      = (blockIdx.x & 1) * 128;
    const int y       = blockIdx.x >> 1;      // 0..255
    const int co_base = blockIdx.y * 128;     // 0..384 step 128

    // staging role: 128 rows x 2 K-halves of 16 ci each
    const int sl   = tid & 127;
    const int half = tid >> 7;

    const float dm = demod[co_base + sl];

    v8f acc[2][4];
    #pragma unroll
    for (int m = 0; m < 2; ++m)
        #pragma unroll
        for (int n = 0; n < 4; ++n)
            acc[m][n] = (v8f){0.f, 0.f, 0.f, 0.f, 0.f, 0.f, 0.f, 0.f};

    // ---- stage K-step s into LDS buffer pb (global fp32 -> hi/lo bf16) ----
    auto stage = [&](int s, int pb) {
        const int tap = s >> 4;                 // 0..8 == kh*3+kw
        const int kh  = (tap * 11) >> 5;        // tap/3
        const int kw  = tap - kh * 3;
        const int ci  = ((s & 15) << 5) + half * 16;
        const int ye  = y + kh - 1;
        const int xg  = x0 + sl + kw - 1;
        const bool bok = ((unsigned)ye < 256u) && ((unsigned)xg < 256u);

        // A: demodulated weights, 128(co) x 32(ci)
        {
            const float* wp = wgt + ((size_t)(co_base + sl) * CIN + ci) * 9 + tap;
            unsigned ph[8], pl[8];
            #pragma unroll
            for (int i = 0; i < 8; ++i) {
                float f0 = wp[(2 * i) * 9]     * dm;
                float f1 = wp[(2 * i + 1) * 9] * dm;
                unsigned u0 = __float_as_uint(f0), u1 = __float_as_uint(f1);
                float l0 = f0 - __uint_as_float(u0 & 0xFFFF0000u);   // exact residual
                float l1 = f1 - __uint_as_float(u1 & 0xFFFF0000u);
                ph[i] = pack_hi16(u1, u0);
                pl[i] = pack_hi16(__float_as_uint(l1), __float_as_uint(l0));
            }
            unsigned short* dh = &ldsA[pb][0][sl * LDSTR + half * 16];
            unsigned short* dl = &ldsA[pb][1][sl * LDSTR + half * 16];
            v4u a0 = {ph[0], ph[1], ph[2], ph[3]};
            v4u a1 = {ph[4], ph[5], ph[6], ph[7]};
            v4u a2 = {pl[0], pl[1], pl[2], pl[3]};
            v4u a3 = {pl[4], pl[5], pl[6], pl[7]};
            ((v4u*)dh)[0] = a0;  ((v4u*)dh)[1] = a1;
            ((v4u*)dl)[0] = a2;  ((v4u*)dl)[1] = a3;
        }
        // B: input patch, 128(x) x 32(ci), zero-padded at borders
        {
            const float* ip = in + (size_t)ci * HW + (size_t)ye * W + xg;
            unsigned ph[8], pl[8];
            #pragma unroll
            for (int i = 0; i < 8; ++i) {
                float f0 = bok ? ip[(size_t)(2 * i) * HW]     : 0.f;
                float f1 = bok ? ip[(size_t)(2 * i + 1) * HW] : 0.f;
                unsigned u0 = __float_as_uint(f0), u1 = __float_as_uint(f1);
                float l0 = f0 - __uint_as_float(u0 & 0xFFFF0000u);
                float l1 = f1 - __uint_as_float(u1 & 0xFFFF0000u);
                ph[i] = pack_hi16(u1, u0);
                pl[i] = pack_hi16(__float_as_uint(l1), __float_as_uint(l0));
            }
            unsigned short* dh = &ldsB[pb][0][sl * LDSTR + half * 16];
            unsigned short* dl = &ldsB[pb][1][sl * LDSTR + half * 16];
            v4u b0 = {ph[0], ph[1], ph[2], ph[3]};
            v4u b1 = {ph[4], ph[5], ph[6], ph[7]};
            v4u b2 = {pl[0], pl[1], pl[2], pl[3]};
            v4u b3 = {pl[4], pl[5], pl[6], pl[7]};
            ((v4u*)dh)[0] = b0;  ((v4u*)dh)[1] = b1;
            ((v4u*)dl)[0] = b2;  ((v4u*)dl)[1] = b3;
        }
    };

    stage(0, 0);                               // prologue

    for (int s = 0; s < NSTEP; ++s) {
        __syncthreads();                       // buf[s&1] writes visible; prior reads done
        const int pb = s & 1;

        // ---- load fragments (ISA 16-bit A 16x32 / B 32x16 lane layout) ----
        FragU ah[2], al[2], bh[4], bl[4];
        #pragma unroll
        for (int m = 0; m < 2; ++m) {
            const int off = (mwave * 32 + m * 16 + (lane & 15)) * LDSTR
                          + ((lane & 16) ? 8 : 0);
            const unsigned short* p0 = &ldsA[pb][0][off];
            const unsigned short* p1 = &ldsA[pb][1][off];
            ah[m].q[0] = *(const v4u*)p0;  ah[m].q[1] = *(const v4u*)(p0 + 16);
            al[m].q[0] = *(const v4u*)p1;  al[m].q[1] = *(const v4u*)(p1 + 16);
        }
        #pragma unroll
        for (int n = 0; n < 4; ++n) {
            const int off = (nwave * 64 + n * 16 + (lane & 15)) * LDSTR
                          + ((lane & 16) ? 8 : 0);
            const unsigned short* p0 = &ldsB[pb][0][off];
            const unsigned short* p1 = &ldsB[pb][1][off];
            bh[n].q[0] = *(const v4u*)p0;  bh[n].q[1] = *(const v4u*)(p0 + 16);
            bl[n].q[0] = *(const v4u*)p1;  bl[n].q[1] = *(const v4u*)(p1 + 16);
        }

        // ---- prefetch next slice into the other buffer (hides VMEM latency) ----
        if (s + 1 < NSTEP) stage(s + 1, pb ^ 1);

        // ---- 3-term split-bf16 MMA: ah*bh + ah*bl + al*bh ----
        #pragma unroll
        for (int m = 0; m < 2; ++m)
            #pragma unroll
            for (int n = 0; n < 4; ++n) {
                acc[m][n] = __builtin_amdgcn_wmma_f32_16x16x32_bf16(
                    false, ah[m].v, false, bh[n].v, (short)0, acc[m][n], false, false);
                acc[m][n] = __builtin_amdgcn_wmma_f32_16x16x32_bf16(
                    false, ah[m].v, false, bl[n].v, (short)0, acc[m][n], false, false);
                acc[m][n] = __builtin_amdgcn_wmma_f32_16x16x32_bf16(
                    false, al[m].v, false, bh[n].v, (short)0, acc[m][n], false, false);
            }
    }

    // ---- epilogue: C/D 16x16 layout -> out[co][y][x] ----
    const int xl  = x0 + nwave * 64 + (lane & 15);
    const int coW = co_base + mwave * 32 + ((lane & 16) ? 8 : 0);
    #pragma unroll
    for (int m = 0; m < 2; ++m)
        #pragma unroll
        for (int n = 0; n < 4; ++n)
            #pragma unroll
            for (int r = 0; r < 8; ++r)
                out[(size_t)(coW + m * 16 + r) * HW + (size_t)y * W + xl + n * 16]
                    = acc[m][n][r];
}

// ---------------------------------------------------------------------------
extern "C" void kernel_launch(void* const* d_in, const int* in_sizes, int n_in,
                              void* d_out, int out_size, void* d_ws, size_t ws_size,
                              hipStream_t stream) {
    const float* in  = (const float*)d_in[0];   // (1,512,256,256) fp32
    const float* wgt = (const float*)d_in[1];   // (1,512,512,3,3) fp32
    float* dm  = (float*)d_ws;                  // 512 floats (2 KB) of scratch
    float* out = (float*)d_out;                 // (1,512,256,256) fp32

    demod512_kernel<<<512, 256, 0, stream>>>(wgt, dm);
    demodconv_wmma_kernel<<<dim3(512, 4), 256, 0, stream>>>(in, wgt, dm, out);
}
